// CAModule_58737972740598
// MI455X (gfx1250) — compile-verified, matrix operand order checked by reading, not run on hardware
//
#include <hip/hip_runtime.h>
#include <hip/hip_bf16.h>

// ---------------------------------------------------------------------------
// Channel-attention module on gfx1250 (wave32, WMMA f32<=f16 16x16x32).
// All GEMMs are NT-form: per-lane operand loads are two contiguous 16B chunks,
// matching the CDNA5 16-bit A/B fragment layout (CDNA5 ISA 7.12.2).
// Each wave computes a 32(M) x 64(N) tile. The K loop (16 steps) is fully
// unrolled so every global load is base+immediate (no loop-carried address
// VALU that would WAR-hazard the WMMA operand registers -> no v_nop padding),
// and B fragments are loaded as a batch before the 8-WMMA burst per step.
// ---------------------------------------------------------------------------

typedef _Float16 v8h  __attribute__((ext_vector_type(8)));
typedef _Float16 v16h __attribute__((ext_vector_type(16)));
typedef float    v8f  __attribute__((ext_vector_type(8)));

static constexpr int NB  = 8;     // batch
static constexpr int NC  = 512;   // channels
static constexpr int NC8 = 64;    // channels/8
static constexpr int NHW = 4096;  // 64*64

// ---- fragment loaders ------------------------------------------------------
// lane L: row = base + (L & 15); khi = L >> 4 selects which K-half the lane
// holds: halves [khi*8 .. khi*8+7] and [16+khi*8 .. 16+khi*8+7] of the 32-wide
// K window -> two 16B loads.
__device__ __forceinline__ v16h load_frag_h(const _Float16* __restrict__ row, int khi) {
    v8h lo = *(const v8h*)(row + khi * 8);
    v8h hi = *(const v8h*)(row + 16 + khi * 8);
    return __builtin_shufflevector(lo, hi, 0,1,2,3,4,5,6,7,8,9,10,11,12,13,14,15);
}

__device__ __forceinline__ v16h load_frag_f(const float* __restrict__ row, int khi) {
    v8f lo = *(const v8f*)(row + khi * 8);
    v8f hi = *(const v8f*)(row + 16 + khi * 8);
    v16h r;
#pragma unroll
    for (int i = 0; i < 8; ++i) { r[i] = (_Float16)lo[i]; r[i + 8] = (_Float16)hi[i]; }
    return r;
}

__device__ __forceinline__ v8f wmma_f16(v16h a, v16h b, v8f c) {
    return __builtin_amdgcn_wmma_f32_16x16x32_f16(
        /*neg_a=*/false, a, /*neg_b=*/false, b,
        /*c_mod=*/(short)0, c, /*reuse_a=*/false, /*reuse_b=*/false);
}

// ---- 1) x (B,C,HW) f32  ->  xT (B,HW,C) f16, LDS-tiled transpose -----------
__global__ __launch_bounds__(256) void k_pack_xT(const float* __restrict__ x,
                                                 _Float16* __restrict__ xT) {
    __shared__ float tile[32][33];
    const int n0 = blockIdx.x * 32;
    const int c0 = blockIdx.y * 32;
    const int b  = blockIdx.z;
    const float* xb = x + (size_t)b * NC * NHW;
    _Float16*  xTb  = xT + (size_t)b * NHW * NC;
#pragma unroll
    for (int j = 0; j < 4; ++j)
        tile[threadIdx.y + 8 * j][threadIdx.x] =
            xb[(size_t)(c0 + threadIdx.y + 8 * j) * NHW + n0 + threadIdx.x];
    __syncthreads();
#pragma unroll
    for (int j = 0; j < 4; ++j)
        xTb[(size_t)(n0 + threadIdx.y + 8 * j) * NC + c0 + threadIdx.x] =
            (_Float16)tile[threadIdx.x][threadIdx.y + 8 * j];
}

// ---- 2) q/k projection: out[m,n] = sum_k W[m,k]*xT[n,k] + bias[m] ----------
// M=64 (blockIdx.y: 2 tiles of 32), N=4096 (64-wide strips, 8 waves/block).
__global__ __launch_bounds__(256) void k_proj_qk(const float* __restrict__ W,
                                                 const float* __restrict__ bias,
                                                 const _Float16* __restrict__ xT,
                                                 _Float16* __restrict__ outh) {
    const int b    = blockIdx.z;
    const int widx = threadIdx.x >> 5;
    const int lane = threadIdx.x & 31;
    const int mrow = lane & 15, khi = lane >> 4;
    const int m_base = blockIdx.y * 32;
    const int n_base = (blockIdx.x * 8 + widx) * 64;

    const _Float16* xTb = xT + (size_t)b * NHW * NC;
    const float* wrow0  = W + (size_t)(m_base + mrow) * NC;
    const float* wrow1  = wrow0 + (size_t)16 * NC;
    const _Float16* brow[4];
#pragma unroll
    for (int t = 0; t < 4; ++t) brow[t] = xTb + (size_t)(n_base + t * 16 + mrow) * NC;

    v8f acc[2][4] = {};
#pragma unroll
    for (int k0 = 0; k0 < NC; k0 += 32) {
        v16h a0 = load_frag_f(wrow0 + k0, khi);
        v16h a1 = load_frag_f(wrow1 + k0, khi);
        v16h bf[4];
#pragma unroll
        for (int t = 0; t < 4; ++t) bf[t] = load_frag_h(brow[t] + k0, khi);
#pragma unroll
        for (int t = 0; t < 4; ++t) {
            acc[0][t] = wmma_f16(a0, bf[t], acc[0][t]);
            acc[1][t] = wmma_f16(a1, bf[t], acc[1][t]);
        }
    }
    _Float16* ob = outh + (size_t)b * NC8 * NHW;
#pragma unroll
    for (int mt = 0; mt < 2; ++mt)
#pragma unroll
        for (int t = 0; t < 4; ++t)
#pragma unroll
            for (int r = 0; r < 8; ++r) {
                const int row = m_base + mt * 16 + r + 8 * khi;
                const int col = n_base + t * 16 + mrow;
                ob[(size_t)row * NHW + col] = (_Float16)(acc[mt][t][r] + bias[row]);
            }
}

// ---- 3) vT[n,d] = sum_k xT[n,k]*Wv[d,k] + bv[d]  (direct transposed V) -----
// M=4096 (n, blockIdx.x: 128 tiles of 32), N=512 (d, 8 strips = 8 waves).
__global__ __launch_bounds__(256) void k_proj_vT(const float* __restrict__ Wv,
                                                 const float* __restrict__ bv,
                                                 const _Float16* __restrict__ xT,
                                                 _Float16* __restrict__ vT) {
    const int b    = blockIdx.z;
    const int widx = threadIdx.x >> 5;
    const int lane = threadIdx.x & 31;
    const int mrow = lane & 15, khi = lane >> 4;
    const int m_base = blockIdx.x * 32;   // n
    const int d_base = widx * 64;         // d

    const _Float16* xTb   = xT + (size_t)b * NHW * NC;
    const _Float16* arow0 = xTb + (size_t)(m_base + mrow) * NC;
    const _Float16* arow1 = arow0 + (size_t)16 * NC;
    const float* brow[4];
#pragma unroll
    for (int t = 0; t < 4; ++t) brow[t] = Wv + (size_t)(d_base + t * 16 + mrow) * NC;

    v8f acc[2][4] = {};
#pragma unroll
    for (int k0 = 0; k0 < NC; k0 += 32) {
        v16h a0 = load_frag_h(arow0 + k0, khi);
        v16h a1 = load_frag_h(arow1 + k0, khi);
        v16h bf[4];
#pragma unroll
        for (int t = 0; t < 4; ++t) bf[t] = load_frag_f(brow[t] + k0, khi);
#pragma unroll
        for (int t = 0; t < 4; ++t) {
            acc[0][t] = wmma_f16(a0, bf[t], acc[0][t]);
            acc[1][t] = wmma_f16(a1, bf[t], acc[1][t]);
        }
    }
    _Float16* vTb = vT + (size_t)b * NHW * NC;
#pragma unroll
    for (int mt = 0; mt < 2; ++mt)
#pragma unroll
        for (int t = 0; t < 4; ++t)
#pragma unroll
            for (int r = 0; r < 8; ++r) {
                const int row = m_base + mt * 16 + r + 8 * khi;   // n
                const int col = d_base + t * 16 + mrow;           // d
                vTb[(size_t)row * NC + col] = (_Float16)(acc[mt][t][r] + bv[col]);
            }
}

// ---- 4) E[i,j] = sum_k qf[i,k]*kf[j,k]  (512x512x512, NT) ------------------
__global__ __launch_bounds__(256) void k_energy(const _Float16* __restrict__ qh,
                                                const _Float16* __restrict__ kh,
                                                float* __restrict__ E) {
    const int b    = blockIdx.z;
    const int widx = threadIdx.x >> 5;
    const int lane = threadIdx.x & 31;
    const int mrow = lane & 15, khi = lane >> 4;
    const int i_base = blockIdx.x * 32;
    const int j_base = widx * 64;

    const _Float16* qb = qh + (size_t)b * NC * NC;   // qf viewed as 512x512
    const _Float16* kb = kh + (size_t)b * NC * NC;
    const _Float16* arow0 = qb + (size_t)(i_base + mrow) * NC;
    const _Float16* arow1 = arow0 + (size_t)16 * NC;
    const _Float16* brow[4];
#pragma unroll
    for (int t = 0; t < 4; ++t) brow[t] = kb + (size_t)(j_base + t * 16 + mrow) * NC;

    v8f acc[2][4] = {};
#pragma unroll
    for (int k0 = 0; k0 < NC; k0 += 32) {
        v16h a0 = load_frag_h(arow0 + k0, khi);
        v16h a1 = load_frag_h(arow1 + k0, khi);
        v16h bf[4];
#pragma unroll
        for (int t = 0; t < 4; ++t) bf[t] = load_frag_h(brow[t] + k0, khi);
#pragma unroll
        for (int t = 0; t < 4; ++t) {
            acc[0][t] = wmma_f16(a0, bf[t], acc[0][t]);
            acc[1][t] = wmma_f16(a1, bf[t], acc[1][t]);
        }
    }
    float* Eb = E + (size_t)b * NC * NC;
#pragma unroll
    for (int mt = 0; mt < 2; ++mt)
#pragma unroll
        for (int t = 0; t < 4; ++t)
#pragma unroll
            for (int r = 0; r < 8; ++r)
                Eb[(size_t)(i_base + mt * 16 + r + 8 * khi) * NC + j_base + t * 16 + mrow] =
                    acc[mt][t][r];
}

// ---- 5) row softmax (fp32), emit attn as f16; one wave per 512-wide row ----
__global__ __launch_bounds__(32) void k_softmax(const float* __restrict__ E,
                                                _Float16* __restrict__ attn) {
    const int row = blockIdx.x, b = blockIdx.y, lane = threadIdx.x;
    const float* er = E + ((size_t)b * NC + row) * NC;
    float v[16];
    float mx = -3.4e38f;
#pragma unroll
    for (int i = 0; i < 16; ++i) { v[i] = er[i * 32 + lane]; mx = fmaxf(mx, v[i]); }
#pragma unroll
    for (int off = 16; off >= 1; off >>= 1) mx = fmaxf(mx, __shfl_xor(mx, off, 32));
    float s = 0.f;
#pragma unroll
    for (int i = 0; i < 16; ++i) { v[i] = __expf(v[i] - mx); s += v[i]; }
#pragma unroll
    for (int off = 16; off >= 1; off >>= 1) s += __shfl_xor(s, off, 32);
    const float inv = 1.f / s;
    _Float16* ar = attn + ((size_t)b * NC + row) * NC;
#pragma unroll
    for (int i = 0; i < 16; ++i) ar[i * 32 + lane] = (_Float16)(v[i] * inv);
}

// ---- 6) out[c,n] = x[c,n] + sum_d attn[c,d]*vT[n,d]  (512x4096x512, NT) ----
__global__ __launch_bounds__(256) void k_out(const _Float16* __restrict__ attn,
                                             const _Float16* __restrict__ vT,
                                             const float* __restrict__ x,
                                             float* __restrict__ out) {
    const int b    = blockIdx.z;
    const int widx = threadIdx.x >> 5;
    const int lane = threadIdx.x & 31;
    const int mrow = lane & 15, khi = lane >> 4;
    const int c_base = blockIdx.y * 32;
    const int n_base = (blockIdx.x * 8 + widx) * 64;

    const _Float16* ab  = attn + (size_t)b * NC * NC;
    const _Float16* vTb = vT + (size_t)b * NHW * NC;
    const _Float16* arow0 = ab + (size_t)(c_base + mrow) * NC;
    const _Float16* arow1 = arow0 + (size_t)16 * NC;
    const _Float16* brow[4];
#pragma unroll
    for (int t = 0; t < 4; ++t) brow[t] = vTb + (size_t)(n_base + t * 16 + mrow) * NC;

    v8f acc[2][4] = {};
#pragma unroll
    for (int k0 = 0; k0 < NC; k0 += 32) {
        v16h a0 = load_frag_h(arow0 + k0, khi);
        v16h a1 = load_frag_h(arow1 + k0, khi);
        v16h bf[4];
#pragma unroll
        for (int t = 0; t < 4; ++t) bf[t] = load_frag_h(brow[t] + k0, khi);
#pragma unroll
        for (int t = 0; t < 4; ++t) {
            acc[0][t] = wmma_f16(a0, bf[t], acc[0][t]);
            acc[1][t] = wmma_f16(a1, bf[t], acc[1][t]);
        }
    }
    const size_t base = (size_t)b * NC * NHW;
#pragma unroll
    for (int mt = 0; mt < 2; ++mt)
#pragma unroll
        for (int t = 0; t < 4; ++t)
#pragma unroll
            for (int r = 0; r < 8; ++r) {
                const size_t idx = base + (size_t)(c_base + mt * 16 + r + 8 * khi) * NHW
                                 + n_base + t * 16 + mrow;
                out[idx] = acc[mt][t][r] + x[idx];
            }
}

// ---------------------------------------------------------------------------
extern "C" void kernel_launch(void* const* d_in, const int* in_sizes, int n_in,
                              void* d_out, int out_size, void* d_ws, size_t ws_size,
                              hipStream_t stream) {
    const float* x  = (const float*)d_in[0];
    const float* Wq = (const float*)d_in[1];
    const float* bq = (const float*)d_in[2];
    const float* Wk = (const float*)d_in[3];
    const float* bk = (const float*)d_in[4];
    const float* Wv = (const float*)d_in[5];
    const float* bv = (const float*)d_in[6];
    float* out = (float*)d_out;

    // workspace carve-out (~88.1 MB total)
    char* ws = (char*)d_ws;
    _Float16* xT   = (_Float16*)(ws);                      // 8*4096*512*2 = 33,554,432
    _Float16* vT   = (_Float16*)(ws + 33554432ull);        // 33,554,432
    _Float16* qh   = (_Float16*)(ws + 67108864ull);        // 8*64*4096*2 =  4,194,304
    _Float16* kh   = (_Float16*)(ws + 71303168ull);        //  4,194,304
    float*    E    = (float*)   (ws + 75497472ull);        // 8*512*512*4 =  8,388,608
    _Float16* attn = (_Float16*)(ws + 83886080ull);        //  4,194,304 -> 88,080,384

    // 1) transpose+pack x -> xT (f16)
    k_pack_xT<<<dim3(NHW / 32, NC / 32, NB), dim3(32, 8), 0, stream>>>(x, xT);

    // 2) q, k projections (M=64, N=4096, K=512), 32x64 tile per wave
    k_proj_qk<<<dim3(8, 2, NB), 256, 0, stream>>>(Wq, bq, xT, qh);
    k_proj_qk<<<dim3(8, 2, NB), 256, 0, stream>>>(Wk, bk, xT, kh);

    // 3) vT = (Wv x + bv)^T directly (M=4096, N=512, K=512)
    k_proj_vT<<<dim3(NHW / 32, 1, NB), 256, 0, stream>>>(Wv, bv, xT, vT);

    // 4) energy E = Qf Kf^T (512x512x512 per batch)
    k_energy<<<dim3(NC / 32, 1, NB), 256, 0, stream>>>(qh, kh, E);

    // 5) softmax rows -> attn (f16)
    k_softmax<<<dim3(NC, NB), 32, 0, stream>>>(E, attn);

    // 6) out = x + attn * Vf (512x4096x512 per batch), fused residual
    k_out<<<dim3(8, NC / 32, NB), 256, 0, stream>>>(attn, vT, x, out);
}
